// EHFSSR_61263413510502
// MI455X (gfx1250) — matrix operand order repaired and checked by reading, not compiled, not run on hardware
//
#include <hip/hip_runtime.h>

// ---------------------------------------------------------------------------
// EHFSSR stereo cross-attention block for gfx1250 (MI455X), bf16 WMMA pipeline
// B=2, C=128, H=128, W=256, NW=4, CG=32, MAXDISP=96
// Intermediates: pixel-major bf16 [b][p][c] for GEMM/attn A,B-of-S operands;
// channel-major bf16 [b][c][p] for attention V operands. Weights staged bf16.
// ---------------------------------------------------------------------------

typedef __attribute__((ext_vector_type(16))) __bf16 v16bf;
typedef __attribute__((ext_vector_type(8)))  float  v8f;
typedef __bf16 bf16_t;

#define CB   128
#define HB   128
#define WB   256
#define HWB  32768          // H*W
#define NTOT 8388608        // B*C*H*W
#define MAXD 96

union frag16 { uint4 q[2]; v16bf v; };
union pack8  { uint4 q;    bf16_t h[8]; };

// ---------------------------------------------------------------------------
// K0: weight f32 -> bf16 staging (128x128)
// ---------------------------------------------------------------------------
__global__ __launch_bounds__(256) void wcvt_kernel(
    const float* __restrict__ src, bf16_t* __restrict__ dst)
{
    int i = blockIdx.x * 256 + threadIdx.x;        // [0, 16384)
    dst[i] = (bf16_t)src[i];
}

// ---------------------------------------------------------------------------
// K1: depthwise 3x3 conv + bias, f32 CM in -> bf16 PM out
// ---------------------------------------------------------------------------
__global__ __launch_bounds__(256) void dwconv3_kernel(
    const float* __restrict__ x, const float* __restrict__ tw,
    const float* __restrict__ tb, bf16_t* __restrict__ out)
{
    int idx = blockIdx.x * 256 + threadIdx.x;      // [0, NTOT)
    int w = idx & 255;
    int h = (idx >> 8) & 127;
    int c = (idx >> 15) & 127;
    int b = idx >> 22;
    const float* xb = x + (size_t)((b * CB + c) * HB) * WB;
    float s = tb[c];
#pragma unroll
    for (int ky = 0; ky < 3; ++ky) {
#pragma unroll
        for (int kx = 0; kx < 3; ++kx) {
            int hs = h + ky - 1, wsx = w + kx - 1;
            if (hs >= 0 && hs < HB && wsx >= 0 && wsx < WB)
                s += tw[c * 9 + ky * 3 + kx] * xb[hs * WB + wsx];
        }
    }
    out[(size_t)(b * HWB + h * WB + w) * CB + c] = (bf16_t)s;
}

// ---------------------------------------------------------------------------
// K2/K4/K6: 128x128 channel GEMM over pixels via v_wmma_f32_16x16x32_bf16.
// A = bf16 weights (staged), B = activations. Wave computes 16(o) x 64(p).
// IN_PM : 1 = pixel-major bf16 (2x b128 per fragment), 0 = channel-major f32.
// OUT_MODE: 0 = CM f32 (d_out), 1 = CM bf16 (V tensors), 2 = PM bf16.
// shifts() fused into B loads (shift group uniform per fragment after unroll).
// ---------------------------------------------------------------------------
template <int SHIFT, int IN_PM, int HAS_BIAS, int HAS_RES, int OUT_MODE>
__global__ __launch_bounds__(256) void gemm128_kernel(
    const bf16_t* __restrict__ Wb, const void* __restrict__ in,
    const float* __restrict__ bias, const float* __restrict__ resid,
    void* __restrict__ out)
{
    const int lane = threadIdx.x & 31;
    const int half = lane >> 4;
    const int lm   = lane & 15;
    const int wid  = blockIdx.x * 8 + (threadIdx.x >> 5);
    const int o0   = (wid & 7) * 16;
    const int p0   = (wid >> 3) * 64;
    const int b    = p0 >> 15;          // HW = 32768
    const int pimg = p0 & 32767;
    const int aoff = half * 16;

    const bf16_t* inb = (const bf16_t*)in;
    const float*  inf = (const float*)in;

    v8f acc[4] = {};

#pragma unroll
    for (int kk = 0; kk < 4; ++kk) {
        // A fragment: bf16 weights, rows o = o0+lm, K slice kk*32..+32
        frag16 ua;
        {
            const bf16_t* wrow = Wb + (o0 + lm) * CB + kk * 32 + aoff;
            ua.q[0] = *(const uint4*)wrow;          // K = aoff .. aoff+7
            ua.q[1] = *(const uint4*)(wrow + 8);    // K = aoff+8 .. aoff+15
        }
        const int i0 = kk * 32 + half * 16;         // first channel of B frag
#pragma unroll
        for (int n = 0; n < 4; ++n) {
            const int pj = pimg + n * 16 + lm;      // pixel within image
            const int hh = pj >> 8, ww = pj & 255;
            // shift group g = kk*2 + half spans the whole fragment
            int hs = hh, wsx = ww; bool z = false;
            if (SHIFT) {
                if (kk == 0) {          // groups 0 (w+1) and 1 (w-1)
                    wsx = half ? (ww - 1) : (ww + 1);
                    z   = half ? (wsx < 0) : (wsx >= WB);
                    wsx = min(max(wsx, 0), WB - 1);
                } else if (kk == 1) {   // groups 2 (h+1) and 3 (h-1)
                    hs = half ? (hh - 1) : (hh + 1);
                    z  = half ? (hs < 0) : (hs >= HB);
                    hs = min(max(hs, 0), HB - 1);
                }                       // kk>=2: identity
            }
            v16bf bb;
            if (IN_PM) {
                const int px = hs * WB + wsx;
                const bf16_t* src = inb + (size_t)(b * HWB + px) * CB + i0;
                frag16 ub;
                ub.q[0] = *(const uint4*)src;
                ub.q[1] = *(const uint4*)(src + 8);
                if (SHIFT && z) { ub.q[0] = {0,0,0,0}; ub.q[1] = {0,0,0,0}; }
                bb = ub.v;
            } else {                    // channel-major f32 (raw inputs)
                const int base = ((b * CB + i0) * HB + hs) * WB + wsx;
#pragma unroll
                for (int e = 0; e < 16; ++e) {
                    float v = (SHIFT && z) ? 0.f : inf[base + e * HWB];
                    bb[e] = (__bf16)v;
                }
            }
            acc[n] = __builtin_amdgcn_wmma_f32_16x16x32_bf16(
                         false, ua.v, false, bb, (short)0, acc[n], false, false);
        }
    }

    // epilogue (C layout: elem r -> M=r+8*half, N=lm)
#pragma unroll
    for (int n = 0; n < 4; ++n) {
        const int pj = pimg + n * 16 + lm;
        if (OUT_MODE == 2) {            // pixel-major bf16, packed b128 store
            pack8 st;
#pragma unroll
            for (int r = 0; r < 8; ++r) {
                const int oo = o0 + r + half * 8;
                float v = acc[n][r];
                if (HAS_BIAS) v += bias[oo];
                if (HAS_RES)  v += resid[(b * CB + oo) * HWB + pj];
                st.h[r] = (bf16_t)v;
            }
            *(uint4*)((bf16_t*)out + (size_t)(b * HWB + pj) * CB + o0 + half * 8) = st.q;
        } else {                        // channel-major scatter
#pragma unroll
            for (int r = 0; r < 8; ++r) {
                const int oo  = o0 + r + half * 8;
                const int off = (b * CB + oo) * HWB + pj;
                float v = acc[n][r];
                if (HAS_BIAS) v += bias[oo];
                if (HAS_RES)  v += resid[off];
                if (OUT_MODE == 1) ((bf16_t*)out)[off] = (bf16_t)v;
                else               ((float*)out)[off]  = v;
            }
        }
    }
}

// ---------------------------------------------------------------------------
// K3: channel LayerNorm per pixel (biased variance), bf16 PM in/out.
// Each thread owns one pixel = 128 contiguous bf16 (256B streaming).
// ---------------------------------------------------------------------------
__global__ __launch_bounds__(256) void ln_kernel(
    const bf16_t* __restrict__ y, const float* __restrict__ lw,
    const float* __restrict__ lb, bf16_t* __restrict__ out)
{
    int pid = blockIdx.x * 256 + threadIdx.x;   // [0, B*HW)
    const bf16_t* base = y + (size_t)pid * CB;
    float s = 0.f, s2 = 0.f;
#pragma unroll 8
    for (int c = 0; c < CB; ++c) {
        float v = (float)base[c];
        s += v; s2 += v * v;
    }
    float mu  = s * (1.f / CB);
    float var = s2 * (1.f / CB) - mu * mu;
    float inv = rsqrtf(var + 1e-6f);
    bf16_t* ob = out + (size_t)pid * CB;
#pragma unroll 8
    for (int c = 0; c < CB; ++c) {
        float v = (float)base[c];
        ob[c] = (bf16_t)(lw[c] * ((v - mu) * inv) + lb[c]);
    }
}

// ---------------------------------------------------------------------------
// K5: banded cross-attention, flash style. One wave per (b,k,h, 16-row block).
// q tensors pixel-major (contiguous channel loads), V channel-major
// (contiguous column loads). Masked logits are 0 (multiplicative mask);
// softmax dense over all 256 columns. Online softmax with shfl_xor reductions
// inside 16-lane halves; P staged through LDS (s_wait_dscnt) to re-layout
// C fragments into an A fragment; P(16x32) x V(32x16) x2 per column pair.
// DIR=0: rows=w (ql), cols=v (qr), V=vr -> warpL. DIR=1: transposed.
// ---------------------------------------------------------------------------
template <int DIR>
__global__ __launch_bounds__(256) void attn_kernel(
    const bf16_t* __restrict__ qrow, const bf16_t* __restrict__ qcol,
    const bf16_t* __restrict__ V, bf16_t* __restrict__ out)
{
    __shared__ __align__(16) bf16_t pbuf[8][16][32];
    const int lane  = threadIdx.x & 31;
    const int half  = lane >> 4;
    const int lm    = lane & 15;
    const int wslot = threadIdx.x >> 5;
    const int wid   = blockIdx.x * 8 + wslot;
    const int rb = wid & 15;               // 16-row block (W/16)
    const int hh = (wid >> 4) & 127;
    const int k  = (wid >> 11) & 3;
    const int b  = wid >> 13;
    const int cb2   = k * 32;              // channel base within C
    const int cbase = b * CB + cb2;        // for channel-major V
    const int aoff  = half * 16;

    // pixel-major offset of (pixel x, channel cb2+c)
    auto QX = [&](int x) { return (size_t)(b * HWB + hh * WB + x) * CB + cb2; };
    // channel-major offset of (channel cb2+c, pixel x)
    auto AT = [&](int c, int x) { return ((cbase + c) * HB + hh) * WB + x; };

    // A fragment: rows i = rb*16 + lm over 32 channels (two 16B chunks)
    frag16 ua;
    {
        const bf16_t* qa = qrow + QX(rb * 16 + lm) + aoff;
        ua.q[0] = *(const uint4*)qa;
        ua.q[1] = *(const uint4*)(qa + 8);
    }

    float m8[8], l8[8];
#pragma unroll
    for (int r = 0; r < 8; ++r) { m8[r] = -3.0e38f; l8[r] = 0.f; }
    v8f acc0 = {}, acc1 = {};

    for (int jp = 0; jp < 8; ++jp) {          // pairs of 16-column tiles
        const int j0 = jp * 32;
        // qcol B fragments: lane = column j, 16 contiguous channels
        frag16 ub0, ub1;
        {
            const bf16_t* q0 = qcol + QX(j0 + lm) + half * 16;
            const bf16_t* q1 = qcol + QX(j0 + 16 + lm) + half * 16;
            ub0.q[0] = *(const uint4*)q0; ub0.q[1] = *(const uint4*)(q0 + 8);
            ub1.q[0] = *(const uint4*)q1; ub1.q[1] = *(const uint4*)(q1 + 8);
        }
        v8f zero = {};
        v8f s0 = __builtin_amdgcn_wmma_f32_16x16x32_bf16(
                     false, ua.v, false, ub0.v, (short)0, zero, false, false);
        v8f s1 = __builtin_amdgcn_wmma_f32_16x16x32_bf16(
                     false, ua.v, false, ub1.v, (short)0, zero, false, false);

        float p0v[8], p1v[8];
#pragma unroll
        for (int r = 0; r < 8; ++r) {
            const int i  = rb * 16 + r + half * 8;
            const int ja = j0 + lm, jb = ja + 16;
            float sa = s0[r], sb = s1[r];
            { int d = DIR ? (ja - i) : (i - ja); if (d < 0 || d > MAXD) sa = 0.f; }
            { int d = DIR ? (jb - i) : (i - jb); if (d < 0 || d > MAXD) sb = 0.f; }
            // row max across the 16 lanes of this half
            float tm = fmaxf(sa, sb);
            for (int off = 1; off < 16; off <<= 1)
                tm = fmaxf(tm, __shfl_xor(tm, off, 32));
            float mn = fmaxf(m8[r], tm);
            float sc = __expf(m8[r] - mn);
            float pa = __expf(sa - mn), pb = __expf(sb - mn);
            float rs = pa + pb;
            for (int off = 1; off < 16; off <<= 1)
                rs += __shfl_xor(rs, off, 32);
            l8[r]  = l8[r] * sc + rs;
            acc0[r] *= sc;
            acc1[r] *= sc;
            m8[r] = mn;
            p0v[r] = pa; p1v[r] = pb;
        }

        // stage P (C layout) -> LDS -> reload as A fragment (16 x 32 over j)
#pragma unroll
        for (int r = 0; r < 8; ++r) {
            const int row = r + half * 8;
            pbuf[wslot][row][lm]      = (bf16_t)p0v[r];
            pbuf[wslot][row][16 + lm] = (bf16_t)p1v[r];
        }
        asm volatile("s_wait_dscnt 0" ::: "memory");
        frag16 up;
        {
            const bf16_t* prow = &pbuf[wslot][lm][0];
            up.q[0] = *(const uint4*)(prow + aoff);
            up.q[1] = *(const uint4*)(prow + aoff + 8);
        }

        // V B fragments (channel-major): lane = channel c, contiguous columns
        frag16 uv0, uv1;
        {
            const bf16_t* vr0 = V + AT(lm,      j0 + half * 16);
            const bf16_t* vr1 = V + AT(16 + lm, j0 + half * 16);
            uv0.q[0] = *(const uint4*)vr0; uv0.q[1] = *(const uint4*)(vr0 + 8);
            uv1.q[0] = *(const uint4*)vr1; uv1.q[1] = *(const uint4*)(vr1 + 8);
        }
        acc0 = __builtin_amdgcn_wmma_f32_16x16x32_bf16(
                   false, up.v, false, uv0.v, (short)0, acc0, false, false);
        acc1 = __builtin_amdgcn_wmma_f32_16x16x32_bf16(
                   false, up.v, false, uv1.v, (short)0, acc1, false, false);
    }

    // normalize and store pixel-major: out[(b, pixel iRow, channel cb2+c)]
#pragma unroll
    for (int r = 0; r < 8; ++r) {
        const int iRow = rb * 16 + r + half * 8;
        const float invl = 1.f / l8[r];
        bf16_t* ob = out + QX(iRow);
        ob[lm]      = (bf16_t)(acc0[r] * invl);
        ob[16 + lm] = (bf16_t)(acc1[r] * invl);
    }
}

// ---------------------------------------------------------------------------
// Host launch
// ---------------------------------------------------------------------------
extern "C" void kernel_launch(void* const* d_in, const int* in_sizes, int n_in,
                              void* d_out, int out_size, void* d_ws, size_t ws_size,
                              hipStream_t stream)
{
    (void)in_sizes; (void)n_in; (void)out_size; (void)ws_size;
    const float* xl0    = (const float*)d_in[0];
    const float* xr0    = (const float*)d_in[1];
    const float* ln_l_w = (const float*)d_in[2];
    const float* ln_l_b = (const float*)d_in[3];
    const float* ln_r_w = (const float*)d_in[4];
    const float* ln_r_b = (const float*)d_in[5];
    const float* feaL_w = (const float*)d_in[6];
    const float* feaR_w = (const float*)d_in[7];
    const float* to_l_w = (const float*)d_in[8];
    const float* to_r_w = (const float*)d_in[9];
    const float* td_w   = (const float*)d_in[10];
    const float* td_b   = (const float*)d_in[11];
    const float* tp_w   = (const float*)d_in[12];
    const float* tp_b   = (const float*)d_in[13];
    const float* out_w  = (const float*)d_in[14];

    char* ws = (char*)d_ws;
    const size_t SEG = 16ull << 20;              // 16 MiB per bf16 tensor
    bf16_t* tmp1 = (bf16_t*)(ws + 0 * SEG);      // PM: dwconv out / warpL
    bf16_t* tmp2 = (bf16_t*)(ws + 1 * SEG);      // PM: transition out / warpR
    bf16_t* xlnL = (bf16_t*)(ws + 2 * SEG);      // PM
    bf16_t* xlnR = (bf16_t*)(ws + 3 * SEG);      // PM
    bf16_t* qlb  = (bf16_t*)(ws + 4 * SEG);      // PM
    bf16_t* qrb  = (bf16_t*)(ws + 5 * SEG);      // PM
    bf16_t* vlb  = (bf16_t*)(ws + 6 * SEG);      // CM
    bf16_t* vrb  = (bf16_t*)(ws + 7 * SEG);      // CM
    char* wsw = ws + 8 * SEG;                    // staged bf16 weights (32KB ea)
    bf16_t* wTP = (bf16_t*)(wsw + 0 * 32768);
    bf16_t* wFL = (bf16_t*)(wsw + 1 * 32768);
    bf16_t* wFR = (bf16_t*)(wsw + 2 * 32768);
    bf16_t* wTL = (bf16_t*)(wsw + 3 * 32768);
    bf16_t* wTR = (bf16_t*)(wsw + 4 * 32768);
    bf16_t* wOU = (bf16_t*)(wsw + 5 * 32768);
    bf16_t* warpL = tmp1;                        // transition temps are dead
    bf16_t* warpR = tmp2;
    float* outL = (float*)d_out;
    float* outR = outL + NTOT;

    // stage weights to bf16
    wcvt_kernel<<<64, 256, 0, stream>>>(tp_w,   wTP);
    wcvt_kernel<<<64, 256, 0, stream>>>(feaL_w, wFL);
    wcvt_kernel<<<64, 256, 0, stream>>>(feaR_w, wFR);
    wcvt_kernel<<<64, 256, 0, stream>>>(to_l_w, wTL);
    wcvt_kernel<<<64, 256, 0, stream>>>(to_r_w, wTR);
    wcvt_kernel<<<64, 256, 0, stream>>>(out_w,  wOU);

    // left transition + LN
    dwconv3_kernel<<<32768, 256, 0, stream>>>(xl0, td_w, td_b, tmp1);
    gemm128_kernel<0,1,1,1,2><<<1024, 256, 0, stream>>>(wTP, tmp1, tp_b, xl0, tmp2);
    ln_kernel<<<256, 256, 0, stream>>>(tmp2, ln_l_w, ln_l_b, xlnL);
    // right transition + LN
    dwconv3_kernel<<<32768, 256, 0, stream>>>(xr0, td_w, td_b, tmp1);
    gemm128_kernel<0,1,1,1,2><<<1024, 256, 0, stream>>>(wTP, tmp1, tp_b, xr0, tmp2);
    ln_kernel<<<256, 256, 0, stream>>>(tmp2, ln_r_w, ln_r_b, xlnR);
    // shifted projections: values (f32 CM in -> CM bf16), queries (PM -> PM)
    gemm128_kernel<1,0,0,0,1><<<1024, 256, 0, stream>>>(wFL, xl0,  nullptr, nullptr, vlb);
    gemm128_kernel<1,0,0,0,1><<<1024, 256, 0, stream>>>(wFR, xr0,  nullptr, nullptr, vrb);
    gemm128_kernel<1,1,0,0,2><<<1024, 256, 0, stream>>>(wTL, xlnL, nullptr, nullptr, qlb);
    gemm128_kernel<1,1,0,0,2><<<1024, 256, 0, stream>>>(wTR, xlnR, nullptr, nullptr, qrb);
    // cross-view attention (row softmax -> warpL, col softmax -> warpR)
    attn_kernel<0><<<2048, 256, 0, stream>>>(qlb, qrb, vrb, warpL);
    attn_kernel<1><<<2048, 256, 0, stream>>>(qrb, qlb, vlb, warpR);
    // output projection + residual (PM bf16 in -> CM f32 out)
    gemm128_kernel<0,1,0,1,0><<<1024, 256, 0, stream>>>(wOU, warpL, nullptr, xl0, outL);
    gemm128_kernel<0,1,0,1,0><<<1024, 256, 0, stream>>>(wOU, warpR, nullptr, xr0, outR);
}
